// HybridModel_8M_41497974014149
// MI455X (gfx1250) — matrix operand order repaired and checked by reading, not compiled, not run on hardware
//
#include <hip/hip_runtime.h>
#include <math.h>

// ---------------------------------------------------------------------------
// Model dims (from reference)
// ---------------------------------------------------------------------------
#define L_LAYERS 8
#define D_MODEL  384
#define H_HEADS  6
#define K_HEAD   64
#define DI_DIM   768
#define MH_HEADS 12
#define HD_DIM   64
#define DS_DIM   64
#define CK_CONV  4
#define V_VOCAB  32000
#define T_SEQ    1024
#define ZX_W     (2*DI_DIM + 2*DS_DIM + MH_HEADS)   // 1676
#define XBC_W    (DI_DIM + 2*DS_DIM)                // 896
#define FF_W     (4*D_MODEL)                        // 1536

// ---------------------------------------------------------------------------
// WMMA / vector types (gfx1250, wave32)
// ---------------------------------------------------------------------------
typedef __attribute__((ext_vector_type(16))) _Float16 v16h;
typedef __attribute__((ext_vector_type(8)))  _Float16 v8h;
typedef __attribute__((ext_vector_type(4)))  _Float16 v4h;
typedef __attribute__((ext_vector_type(8)))  float    v8f;
typedef __attribute__((ext_vector_type(4)))  float    v4f;

union FragU { v16h v; v8h h[2]; };

// async global->LDS copy of one f32 per lane (VGLOBAL GLOBAL_LOAD_ASYNC_TO_LDS_B32,
// GV mode: VDST = LDS byte address, VADDR = 64-bit global address). Tracked by
// ASYNCcnt; pair with s_wait_asynccnt before the barrier.
__device__ __forceinline__ void async_copy_f32_to_lds(const float* gsrc, void* lds_dst) {
  unsigned ldst = (unsigned)(size_t)lds_dst;                 // low 32 bits = LDS offset
  unsigned long long ga = (unsigned long long)(size_t)gsrc;
  asm volatile("global_load_async_to_lds_b32 %0, %1, off"
               :: "v"(ldst), "v"(ga) : "memory");
}
__device__ __forceinline__ void wait_async_lds() {
#if __has_builtin(__builtin_amdgcn_s_wait_asynccnt)
  __builtin_amdgcn_s_wait_asynccnt(0);
#else
  asm volatile("s_wait_asynccnt 0" ::: "memory");
#endif
}

// ---------------------------------------------------------------------------
// GEMM: C[M,N] = act(A[M,K] @ W + bias)  (f16 WMMA compute, f32 accumulate)
//   WT=0: W is [K,N] row-major;  WT=1: W is [N,K] row-major (i.e. W^T)
//   ACT:  0 none, 1 gelu(exact erf), 2 exp(-exp(x)) with bias (RWKV decay)
//   ACC:  1 -> C += result
// Assumptions baked in (true for every call in this model):
//   M % 128 == 0, K % 32 == 0, N % 4 == 0.  NG=1 adds the N-edge guard.
// Block tile 128x128, BK=32, 256 threads = 8 waves (4 M-slabs x 2 N-slabs),
// each wave: 2x4 grid of v_wmma_f32_16x16x32_f16.
// 2-deep software pipeline: global loads of stage s+1 are issued into
// registers while the WMMAs of stage s run; convert+LDS-store happens after
// the barrier. Removes per-chunk s_wait_loadcnt serialization.
// ---------------------------------------------------------------------------
#define BM 128
#define BN 128
#define BK 32
#define LDPAD 8

template<int WT, int NG, int ACT, int ACC>
__global__ __launch_bounds__(256) void k_gemm(
    const float* __restrict__ A, const float* __restrict__ W,
    const float* __restrict__ bias, float* __restrict__ C,
    int M, int N, int Kd)
{
  __shared__ __align__(16) _Float16 As[BM][BK + LDPAD];
  __shared__ __align__(16) _Float16 Bs[BN][BK + LDPAD];

  const int tid   = threadIdx.x;
  const int mBase = blockIdx.y * BM;
  const int nBase = blockIdx.x * BN;
  const int wave  = tid >> 5;
  const int wm    = wave & 3;    // 32-row slab within block tile
  const int wn    = wave >> 2;   // 64-col slab within block tile
  const int lane  = tid & 31;
  const int lrow  = lane & 15;   // M-row (A frag) / N-col (B frag)
  const int lhi   = lane >> 4;   // half-wave K-group select (ISA 7.12.2)

  v8f acc[2][4];
#pragma unroll
  for (int mi = 0; mi < 2; ++mi)
#pragma unroll
    for (int ni = 0; ni < 4; ++ni)
      acc[mi][ni] = (v8f)(0.f);

  // staging registers: 4 A-chunks + 4 B-chunks of float4 per thread
  v4f aReg[4], bReg[4];

  // issue all global loads of one k-stage back-to-back (no LDS dependence)
  auto loadStage = [&](int k0) {
#pragma unroll
    for (int it = 0; it < 4; ++it) {
      int idx = (tid + it * 256) * 4;
      int r = idx >> 5, c = idx & (BK - 1);
      aReg[it] = *(const v4f*)(A + (size_t)(mBase + r) * Kd + k0 + c);
    }
    if (WT == 0) {
#pragma unroll
      for (int it = 0; it < 4; ++it) {
        int idx = (tid + it * 256) * 4;
        int n = idx & (BN - 1), c = idx >> 7;
        if (!NG || (nBase + n) < N)   // N % 4 == 0 -> chunk all-valid or all-invalid
          bReg[it] = *(const v4f*)(W + (size_t)(k0 + c) * N + nBase + n);
        else
          bReg[it] = (v4f)(0.f);
      }
    } else {
#pragma unroll
      for (int it = 0; it < 4; ++it) {
        int idx = (tid + it * 256) * 4;
        int n = idx >> 5, c = idx & (BK - 1);
        if (!NG || (nBase + n) < N)
          bReg[it] = *(const v4f*)(W + (size_t)(nBase + n) * Kd + k0 + c);
        else
          bReg[it] = (v4f)(0.f);
      }
    }
  };

  // convert f32->f16 and store the staged registers into LDS
  auto storeStage = [&]() {
#pragma unroll
    for (int it = 0; it < 4; ++it) {
      int idx = (tid + it * 256) * 4;
      int r = idx >> 5, c = idx & (BK - 1);
      v4f fv = aReg[it];
      v4h hv = { (_Float16)fv[0], (_Float16)fv[1], (_Float16)fv[2], (_Float16)fv[3] };
      *(v4h*)&As[r][c] = hv;
    }
    if (WT == 0) {
#pragma unroll
      for (int it = 0; it < 4; ++it) {
        int idx = (tid + it * 256) * 4;
        int n = idx & (BN - 1), c = idx >> 7;
        v4f fv = bReg[it];
        Bs[n + 0][c] = (_Float16)fv[0];   // mini-transpose into Bs[n][k]
        Bs[n + 1][c] = (_Float16)fv[1];
        Bs[n + 2][c] = (_Float16)fv[2];
        Bs[n + 3][c] = (_Float16)fv[3];
      }
    } else {
#pragma unroll
      for (int it = 0; it < 4; ++it) {
        int idx = (tid + it * 256) * 4;
        int n = idx >> 5, c = idx & (BK - 1);
        v4f fv = bReg[it];
        v4h hv = { (_Float16)fv[0], (_Float16)fv[1], (_Float16)fv[2], (_Float16)fv[3] };
        *(v4h*)&Bs[n][c] = hv;
      }
    }
  };

  const int nStages = Kd / BK;
  loadStage(0);                               // pipeline prologue
  for (int s = 0; s < nStages; ++s) {
    storeStage();
    __syncthreads();
    if (s + 1 < nStages) loadStage((s + 1) * BK);   // overlap with WMMAs below

    // ---- fragments: two ds_load_b128 each (ISA 7.12.2 16-bit layouts) ----
    v16h af[2], bfv[4];
#pragma unroll
    for (int mi = 0; mi < 2; ++mi) {
      int r = wm * 32 + mi * 16 + lrow;
      FragU f;
      f.h[0] = *(const v8h*)&As[r][8 * lhi];        // K = 0..7   / 8..15
      f.h[1] = *(const v8h*)&As[r][16 + 8 * lhi];   // K = 16..23 / 24..31
      af[mi] = f.v;
    }
#pragma unroll
    for (int ni = 0; ni < 4; ++ni) {
      int cc = wn * 64 + ni * 16 + lrow;
      FragU f;
      f.h[0] = *(const v8h*)&Bs[cc][8 * lhi];       // K = 0..7   / 8..15
      f.h[1] = *(const v8h*)&Bs[cc][16 + 8 * lhi];  // K = 16..23 / 24..31
      bfv[ni] = f.v;
    }
#pragma unroll
    for (int mi = 0; mi < 2; ++mi)
#pragma unroll
      for (int ni = 0; ni < 4; ++ni)
        acc[mi][ni] = __builtin_amdgcn_wmma_f32_16x16x32_f16(
            false, af[mi], false, bfv[ni], (short)0, acc[mi][ni], false, false);
    __syncthreads();
  }

  // ---- epilogue: D layout VGPR r -> M = r + 8*lhi, N = lrow ----
#pragma unroll
  for (int mi = 0; mi < 2; ++mi) {
#pragma unroll
    for (int ni = 0; ni < 4; ++ni) {
      int n = nBase + wn * 64 + ni * 16 + lrow;
      if (NG && n >= N) continue;
      float bv = (ACT == 2) ? bias[n] : 0.f;
#pragma unroll
      for (int r = 0; r < 8; ++r) {
        int m = mBase + wm * 32 + mi * 16 + 8 * lhi + r;
        float v = acc[mi][ni][r] + bv;
        if (ACT == 1)      v = 0.5f * v * (1.f + erff(v * 0.70710678f));
        else if (ACT == 2) v = expf(-expf(v));
        if (ACC) C[(size_t)m * N + n] += v;
        else     C[(size_t)m * N + n] = v;
      }
    }
  }
}

// ---------------------------------------------------------------------------
// Elementwise / normalization / scan kernels
// ---------------------------------------------------------------------------
__global__ __launch_bounds__(256) void k_embed(
    const int* __restrict__ x, const float* __restrict__ embed,
    float* __restrict__ h)
{
  int i = blockIdx.x * 256 + threadIdx.x;
  if (i >= T_SEQ * D_MODEL) return;
  int t = i / D_MODEL, d = i % D_MODEL;
  h[i] = embed[(size_t)x[t] * D_MODEL + d];
}

__global__ __launch_bounds__(256) void k_rmsnorm(
    const float* __restrict__ in, const float* __restrict__ w,
    float* __restrict__ out, int Dim, float eps)
{
  int t = blockIdx.x;
  const float* row = in + (size_t)t * Dim;
  __shared__ float red[8];
  float ss = 0.f;
  for (int i = threadIdx.x; i < Dim; i += 256) { float v = row[i]; ss += v * v; }
  for (int o = 16; o; o >>= 1) ss += __shfl_xor(ss, o, 32);
  if ((threadIdx.x & 31) == 0) red[threadIdx.x >> 5] = ss;
  __syncthreads();
  if (threadIdx.x == 0) {
    float s = 0.f;
    for (int i = 0; i < 8; ++i) s += red[i];
    red[0] = rsqrtf(s / (float)Dim + eps);
  }
  __syncthreads();
  float rinv = red[0];
  for (int i = threadIdx.x; i < Dim; i += 256)
    out[(size_t)t * Dim + i] = w[i] * row[i] * rinv;
}

// RWKV6 scan: one block per head, state S[64][64] striped 16 f32/thread.
// Per-token operands r/k/v/decay are DMA'd global->LDS with async copies.
__global__ __launch_bounds__(256) void k_rwkv_scan(
    const float* __restrict__ r, const float* __restrict__ kk,
    const float* __restrict__ v, const float* __restrict__ dec,
    const float* __restrict__ u, float* __restrict__ o)
{
  int hh = blockIdx.x;
  int tid = threadIdx.x;
  int vv = tid & 63, gg = tid >> 6;
  __shared__ float sin_buf[256];           // [r | k | v | d], 64 each
  __shared__ float part[4][64];
  const float* sr = sin_buf;
  const float* sk = sin_buf + 64;
  const float* sv = sin_buf + 128;
  const float* sd = sin_buf + 192;
  float S[16], uk[16];
  for (int j = 0; j < 16; ++j) { S[j] = 0.f; uk[j] = u[hh * 64 + gg * 16 + j]; }
  for (int t = 0; t < T_SEQ; ++t) {
    int base = t * D_MODEL + hh * 64;
    const float* gsrc = (tid < 64)  ? (r   + base + tid)
                      : (tid < 128) ? (kk  + base + tid - 64)
                      : (tid < 192) ? (v   + base + tid - 128)
                                    : (dec + base + tid - 192);
    async_copy_f32_to_lds(gsrc, (void*)&sin_buf[tid]);
    wait_async_lds();
    __syncthreads();
    float p = 0.f, vval = sv[vv];
#pragma unroll
    for (int j = 0; j < 16; ++j) {
      int ki = gg * 16 + j;
      float kv = sk[ki] * vval;
      p += sr[ki] * (S[j] + uk[j] * kv);
      S[j] = sd[ki] * S[j] + kv;
    }
    part[gg][vv] = p;
    __syncthreads();
    if (gg == 0)
      o[base + vv] = part[0][vv] + part[1][vv] + part[2][vv] + part[3][vv];
    __syncthreads();
  }
}

// GroupNorm(64) + silu(g) gate; one 64-thread block per (t, head).
__global__ __launch_bounds__(64) void k_gn(
    const float* __restrict__ o, const float* __restrict__ g,
    const float* __restrict__ gnw, const float* __restrict__ gnb,
    float* __restrict__ out)
{
  int gidx = blockIdx.x;
  int t = gidx / H_HEADS, hh = gidx % H_HEADS;
  int j = threadIdx.x;
  __shared__ float red[4];
  int base = t * D_MODEL + hh * 64;
  float v = o[base + j];
  float s = v, sq = v * v;
  for (int off = 16; off; off >>= 1) {
    s  += __shfl_xor(s,  off, 32);
    sq += __shfl_xor(sq, off, 32);
  }
  if ((j & 31) == 0) { red[(j >> 5) * 2] = s; red[(j >> 5) * 2 + 1] = sq; }
  __syncthreads();
  float mean = (red[0] + red[2]) * (1.f / 64.f);
  float var  = (red[1] + red[3]) * (1.f / 64.f) - mean * mean;
  float gl   = g[base + j];
  float sil  = gl / (1.f + expf(-gl));
  out[base + j] =
      ((v - mean) * rsqrtf(var + 1e-5f) * gnw[hh * 64 + j] + gnb[hh * 64 + j]) * sil;
}

// dt = softplus(zx[..., -12:] + dt_bias)
__global__ __launch_bounds__(256) void k_dt(
    const float* __restrict__ zx, const float* __restrict__ dtb,
    float* __restrict__ dt)
{
  int i = blockIdx.x * 256 + threadIdx.x;
  if (i >= T_SEQ * MH_HEADS) return;
  int t = i / MH_HEADS, c = i % MH_HEADS;
  float x = zx[(size_t)t * ZX_W + (ZX_W - MH_HEADS) + c] + dtb[c];
  dt[i] = (x > 20.f) ? x : log1pf(expf(x));
}

// causal depthwise conv (CK=4) + bias + silu over the 896 xBC channels
__global__ __launch_bounds__(256) void k_conv(
    const float* __restrict__ zx, const float* __restrict__ cw,
    const float* __restrict__ cb, float* __restrict__ xbc)
{
  int i = blockIdx.x * 256 + threadIdx.x;
  if (i >= T_SEQ * XBC_W) return;
  int t = i / XBC_W, c = i % XBC_W;
  float acc = cb[c];
#pragma unroll
  for (int j = 0; j < CK_CONV; ++j) {
    int tt = t + j - (CK_CONV - 1);
    if (tt >= 0) acc += zx[(size_t)tt * ZX_W + DI_DIM + c] * cw[c * CK_CONV + j];
  }
  xbc[i] = acc / (1.f + expf(-acc));
}

// Mamba2 scan: one block per head, state S[64][64] striped 16 f32/thread.
// Per-token x/B/C are DMA'd global->LDS with async copies.
__global__ __launch_bounds__(256) void k_mamba_scan(
    const float* __restrict__ xbc, const float* __restrict__ dt,
    const float* __restrict__ A_log, const float* __restrict__ Dp,
    float* __restrict__ y)
{
  int hh = blockIdx.x;
  int tid = threadIdx.x;
  int pp = tid & 63, gg = tid >> 6;
  float Ah = -expf(A_log[hh]);
  float Dv = Dp[hh];
  __shared__ float sin_buf[192];           // [x | B | C], 64 each
  __shared__ float part[4][64];
  const float* sx = sin_buf;
  const float* sB = sin_buf + 64;
  const float* sC = sin_buf + 128;
  float S[16];
  for (int j = 0; j < 16; ++j) S[j] = 0.f;
  for (int t = 0; t < T_SEQ; ++t) {
    int xb = t * XBC_W;
    if (tid < 192) {
      const float* gsrc = (tid < 64)  ? (xbc + xb + hh * 64 + tid)
                        : (tid < 128) ? (xbc + xb + DI_DIM + tid - 64)
                                      : (xbc + xb + DI_DIM + DS_DIM + tid - 128);
      async_copy_f32_to_lds(gsrc, (void*)&sin_buf[tid]);
    }
    wait_async_lds();
    __syncthreads();
    float dtv = dt[t * MH_HEADS + hh];
    float dA  = expf(dtv * Ah);
    float dx  = dtv * sx[pp];
    float p = 0.f;
#pragma unroll
    for (int j = 0; j < 16; ++j) {
      int s = gg * 16 + j;
      S[j] = dA * S[j] + dx * sB[s];
      p += S[j] * sC[s];
    }
    part[gg][pp] = p;
    __syncthreads();
    if (gg == 0)
      y[(size_t)t * DI_DIM + hh * 64 + pp] =
          part[0][pp] + part[1][pp] + part[2][pp] + part[3][pp] + Dv * sx[pp];
    __syncthreads();
  }
}

// out = y * silu(z)   (z = zx[..., :768])
__global__ __launch_bounds__(256) void k_mul_silu(
    const float* __restrict__ y, const float* __restrict__ zx,
    float* __restrict__ out)
{
  int i = blockIdx.x * 256 + threadIdx.x;
  if (i >= T_SEQ * DI_DIM) return;
  int t = i / DI_DIM, c = i % DI_DIM;
  float z = zx[(size_t)t * ZX_W + c];
  out[i] = y[i] * z / (1.f + expf(-z));
}

// sigmoid gate over concat(o_r,o_m)@gw + gb; h += gate*o_r + (1-gate)*o_m
__global__ __launch_bounds__(256) void k_gate(
    const float* __restrict__ o_r, const float* __restrict__ o_m,
    const float* __restrict__ gw, const float* __restrict__ gb,
    float* __restrict__ h)
{
  int t = blockIdx.x, tid = threadIdx.x;
  __shared__ float red[8];
  float s = 0.f;
  for (int d = tid; d < D_MODEL; d += 256)
    s += o_r[(size_t)t * D_MODEL + d] * gw[d] +
         o_m[(size_t)t * D_MODEL + d] * gw[D_MODEL + d];
  for (int o = 16; o; o >>= 1) s += __shfl_xor(s, o, 32);
  if ((tid & 31) == 0) red[tid >> 5] = s;
  __syncthreads();
  if (tid == 0) {
    float tt = 0.f;
    for (int i = 0; i < 8; ++i) tt += red[i];
    red[0] = 1.f / (1.f + expf(-(tt + gb[0])));
  }
  __syncthreads();
  float gate = red[0];
  for (int d = tid; d < D_MODEL; d += 256) {
    size_t idx = (size_t)t * D_MODEL + d;
    h[idx] += gate * o_r[idx] + (1.f - gate) * o_m[idx];
  }
}

// ---------------------------------------------------------------------------
// Host launcher
// ---------------------------------------------------------------------------
extern "C" void kernel_launch(void* const* d_in, const int* in_sizes, int n_in,
                              void* d_out, int out_size, void* d_ws, size_t ws_size,
                              hipStream_t stream) {
  const int*   x        = (const int*)  d_in[0];
  const float* embed    = (const float*)d_in[1];
  const float* ln_w     = (const float*)d_in[2];
  const float* Wr       = (const float*)d_in[3];
  const float* Wk       = (const float*)d_in[4];
  const float* Wv       = (const float*)d_in[5];
  const float* Wg       = (const float*)d_in[6];
  const float* Ww       = (const float*)d_in[7];
  const float* w_bias   = (const float*)d_in[8];
  const float* u        = (const float*)d_in[9];
  const float* gn_w     = (const float*)d_in[10];
  const float* gn_b     = (const float*)d_in[11];
  const float* Wo       = (const float*)d_in[12];
  const float* W_in     = (const float*)d_in[13];
  const float* conv_w   = (const float*)d_in[14];
  const float* conv_b   = (const float*)d_in[15];
  const float* dt_bias  = (const float*)d_in[16];
  const float* A_log    = (const float*)d_in[17];
  const float* Dp       = (const float*)d_in[18];
  const float* mn_w     = (const float*)d_in[19];
  const float* W_out    = (const float*)d_in[20];
  const float* gw       = (const float*)d_in[21];
  const float* gb       = (const float*)d_in[22];
  const float* ffn_ln_w = (const float*)d_in[23];
  const float* ffn_w1   = (const float*)d_in[24];
  const float* ffn_w2   = (const float*)d_in[25];
  const float* ln_out_w = (const float*)d_in[26];
  float* out = (float*)d_out;

  // workspace carve (floats)
  float* ws = (float*)d_ws;
  const size_t TD = (size_t)T_SEQ * D_MODEL;
  float* h      = ws;               ws += TD;
  float* xn     = ws;               ws += TD;
  float* bufR   = ws;               ws += TD;
  float* bufK   = ws;               ws += TD;
  float* bufV   = ws;               ws += TD;
  float* bufG   = ws;               ws += TD;
  float* bufW   = ws;               ws += TD;   // decay
  float* bufO   = ws;               ws += TD;   // rwkv scan out
  float* bufAtt = ws;               ws += TD;   // gn*silu(g)
  float* oR     = ws;               ws += TD;
  float* zx     = ws;               ws += (size_t)T_SEQ * ZX_W;
  float* dtb    = ws;               ws += (size_t)T_SEQ * MH_HEADS;
  float* xbc    = ws;               ws += (size_t)T_SEQ * XBC_W;
  float* ybuf   = ws;               ws += (size_t)T_SEQ * DI_DIM;
  float* ytmp   = ws;               ws += (size_t)T_SEQ * DI_DIM;
  float* y2     = ws;               ws += (size_t)T_SEQ * DI_DIM;
  float* oM     = ws;               ws += TD;
  float* hn     = ws;               ws += TD;
  float* mid    = ws;               ws += (size_t)T_SEQ * FF_W;

  auto gemm = [&](const float* A, const float* W, const float* bias, float* C,
                  int M, int N, int K, int wt, int act, int accum) {
    dim3 grid((N + BN - 1) / BN, (M + BM - 1) / BM);
    bool ng = (N % BN) != 0;
    if (wt)            k_gemm<1,0,0,0><<<grid, 256, 0, stream>>>(A, W, bias, C, M, N, K);
    else if (ng)       k_gemm<0,1,0,0><<<grid, 256, 0, stream>>>(A, W, bias, C, M, N, K);
    else if (act == 1) k_gemm<0,0,1,0><<<grid, 256, 0, stream>>>(A, W, bias, C, M, N, K);
    else if (act == 2) k_gemm<0,0,2,0><<<grid, 256, 0, stream>>>(A, W, bias, C, M, N, K);
    else if (accum)    k_gemm<0,0,0,1><<<grid, 256, 0, stream>>>(A, W, bias, C, M, N, K);
    else               k_gemm<0,0,0,0><<<grid, 256, 0, stream>>>(A, W, bias, C, M, N, K);
  };

  // embedding lookup
  k_embed<<<(T_SEQ * D_MODEL) / 256, 256, 0, stream>>>(x, embed, h);

  for (int l = 0; l < L_LAYERS; ++l) {
    const float* Wr_l  = Wr  + (size_t)l * D_MODEL * D_MODEL;
    const float* Wk_l  = Wk  + (size_t)l * D_MODEL * D_MODEL;
    const float* Wv_l  = Wv  + (size_t)l * D_MODEL * D_MODEL;
    const float* Wg_l  = Wg  + (size_t)l * D_MODEL * D_MODEL;
    const float* Ww_l  = Ww  + (size_t)l * D_MODEL * D_MODEL;
    const float* Wo_l  = Wo  + (size_t)l * D_MODEL * D_MODEL;
    const float* Win_l = W_in + (size_t)l * D_MODEL * ZX_W;
    const float* Wout_l= W_out + (size_t)l * DI_DIM * D_MODEL;
    const float* f1_l  = ffn_w1 + (size_t)l * D_MODEL * FF_W;
    const float* f2_l  = ffn_w2 + (size_t)l * FF_W * D_MODEL;

    // ---- RWKV6 branch ----
    k_rmsnorm<<<T_SEQ, 256, 0, stream>>>(h, ln_w + l * D_MODEL, xn, D_MODEL, 1e-6f);
    gemm(xn, Wr_l, nullptr, bufR, T_SEQ, D_MODEL, D_MODEL, 0, 0, 0);
    gemm(xn, Wk_l, nullptr, bufK, T_SEQ, D_MODEL, D_MODEL, 0, 0, 0);
    gemm(xn, Wv_l, nullptr, bufV, T_SEQ, D_MODEL, D_MODEL, 0, 0, 0);
    gemm(xn, Wg_l, nullptr, bufG, T_SEQ, D_MODEL, D_MODEL, 0, 0, 0);
    gemm(xn, Ww_l, w_bias + l * D_MODEL, bufW, T_SEQ, D_MODEL, D_MODEL, 0, 2, 0); // decay fused
    k_rwkv_scan<<<H_HEADS, 256, 0, stream>>>(bufR, bufK, bufV, bufW,
                                             u + (size_t)l * H_HEADS * K_HEAD, bufO);
    k_gn<<<T_SEQ * H_HEADS, 64, 0, stream>>>(bufO, bufG, gn_w + l * D_MODEL,
                                             gn_b + l * D_MODEL, bufAtt);
    gemm(bufAtt, Wo_l, nullptr, oR, T_SEQ, D_MODEL, D_MODEL, 0, 0, 0);

    // ---- Mamba2 branch ----
    gemm(xn, Win_l, nullptr, zx, T_SEQ, ZX_W, D_MODEL, 0, 0, 0);
    k_dt<<<(T_SEQ * MH_HEADS + 255) / 256, 256, 0, stream>>>(zx, dt_bias + l * MH_HEADS, dtb);
    k_conv<<<(T_SEQ * XBC_W) / 256, 256, 0, stream>>>(zx, conv_w + (size_t)l * XBC_W * CK_CONV,
                                                      conv_b + (size_t)l * XBC_W, xbc);
    k_mamba_scan<<<MH_HEADS, 256, 0, stream>>>(xbc, dtb, A_log + l * MH_HEADS,
                                               Dp + l * MH_HEADS, ybuf);
    k_mul_silu<<<(T_SEQ * DI_DIM) / 256, 256, 0, stream>>>(ybuf, zx, ytmp);
    k_rmsnorm<<<T_SEQ, 256, 0, stream>>>(ytmp, mn_w + (size_t)l * DI_DIM, y2, DI_DIM, 1e-6f);
    gemm(y2, Wout_l, nullptr, oM, T_SEQ, D_MODEL, DI_DIM, 0, 0, 0);

    // ---- gated residual merge ----
    k_gate<<<T_SEQ, 256, 0, stream>>>(oR, oM, gw + (size_t)l * 2 * D_MODEL, gb + l, h);

    // ---- FFN ----
    k_rmsnorm<<<T_SEQ, 256, 0, stream>>>(h, ffn_ln_w + l * D_MODEL, hn, D_MODEL, 1e-6f);
    gemm(hn, f1_l, nullptr, mid, T_SEQ, FF_W, D_MODEL, 0, 1, 0);   // gelu fused
    gemm(mid, f2_l, nullptr, h, T_SEQ, D_MODEL, FF_W, 0, 0, 1);    // accumulate residual
  }

  // final norm + logits (W transposed: embed is [V, D])
  k_rmsnorm<<<T_SEQ, 256, 0, stream>>>(h, ln_out_w, xn, D_MODEL, 1e-6f);
  gemm(xn, embed, nullptr, out, T_SEQ, V_VOCAB, D_MODEL, 1, 0, 0);
}
